// InstanceDiscrimination_33139967656579
// MI455X (gfx1250) — matrix operand order repaired and checked by reading, not compiled, not run on hardware
//
#include <hip/hip_runtime.h>
#include <hip/hip_bf16.h>

typedef __attribute__((ext_vector_type(16))) _Float16 v16h;
typedef __attribute__((ext_vector_type(8)))  _Float16 v8h;
typedef __attribute__((ext_vector_type(8)))  float    v8f;
typedef __attribute__((ext_vector_type(4)))  int      v4i;

#define KDIM       128
#define INV_TAU    (1.0f / 0.07f)
#define WAVES_PB   8                    // waves per block
#define ROWS_PB    (WAVES_PB * 16)      // 128 rows per block
#define TROW       136                  // padded LDS row stride in halves (272B -> bank rotate 4)
#define TILE_H     (16 * TROW)          // halves per (hi or lo) tile in LDS

typedef __attribute__((address_space(1))) v4i* gas_v4i;
typedef __attribute__((address_space(3))) v4i* las_v4i;

// ---------------------------------------------------------------------------
// 1) zero workspace accumulators + output scalar
// ---------------------------------------------------------------------------
__global__ void id_zero_kernel(float* __restrict__ den, float* __restrict__ out, int n) {
    int i = blockIdx.x * blockDim.x + threadIdx.x;
    if (i < n) den[i] = 0.0f;
    if (i == 0) out[0] = 0.0f;
}

// ---------------------------------------------------------------------------
// 2) split fp32 -> f16 hi + f16 lo  (v = hi + lo, near-fp32 via 3 WMMAs)
// ---------------------------------------------------------------------------
__global__ void id_split_kernel(const float* __restrict__ v,
                                _Float16* __restrict__ hi,
                                _Float16* __restrict__ lo, int n) {
    int i = blockIdx.x * blockDim.x + threadIdx.x;
    if (i < n) {
        float x = v[i];
        _Float16 h = (_Float16)x;
        hi[i] = h;
        lo[i] = (_Float16)(x - (float)h);
    }
}

// ---------------------------------------------------------------------------
// async-stage one B tile (16 rows of hi + lo) into LDS buffer.
// 512 x 16B chunks total, 256 threads -> 2 async b128 loads per thread
// (= 2 instructions per wave, tracked by ASYNCcnt).
// ---------------------------------------------------------------------------
__device__ __forceinline__ void stage_tile(const _Float16* __restrict__ hi,
                                           const _Float16* __restrict__ lo,
                                           int j0, _Float16* sbuf, int tid) {
    #pragma unroll
    for (int u = 0; u < 2; ++u) {
        const int ci     = tid + u * 256;          // 0..511
        const int mat    = ci >> 8;                // 0 = hi, 1 = lo
        const int within = ci & 255;               // chunk within tile
        const int row    = within >> 4;            // 0..15
        const int seg    = within & 15;            // 16B segment in row
        const _Float16* g = (mat ? lo : hi) + (long)(j0 + row) * KDIM + seg * 8;
        _Float16*       l = sbuf + mat * TILE_H + row * TROW + seg * 8;
        __builtin_amdgcn_global_load_async_to_lds_b128((gas_v4i)g, (las_v4i)l, 0, 0);
    }
}

// ---------------------------------------------------------------------------
// 3) Gram-matrix row-sum kernel.
//    Block = 8 waves; wave w owns rows [blockIdx.x*128 + w*16, +16).
//    B tiles staged once per block into LDS (async, double-buffered),
//    consumed by all 8 waves -> 8x less L2 traffic per WMMA.
// ---------------------------------------------------------------------------
__global__ __launch_bounds__(256) void id_gram_kernel(
    const _Float16* __restrict__ hi, const _Float16* __restrict__ lo,
    float* __restrict__ den, int n_jtiles, int jchunk) {

    __shared__ __align__(16) _Float16 smem[2][2 * TILE_H];   // 2 buffers x (hi,lo)

    const int tid       = threadIdx.x;
    const int lane      = tid & 31;
    const int wave      = tid >> 5;
    const int half_lane = lane & 15;
    const int hsel      = lane >> 4;
    const int i0        = (blockIdx.x * WAVES_PB + wave) * 16;
    const int jt0       = blockIdx.y * jchunk;

    // ---- hoist A tiles (4 k-chunks, hi & lo) into registers ----
    v16h a_hi[4], a_lo[4];
    {
        const long rowA = (long)(i0 + half_lane) * KDIM;
        #pragma unroll
        for (int c = 0; c < 4; ++c) {
            const long base = rowA + c * 32 + hsel * 8;
            v8h h0 = *(const v8h*)(hi + base);
            v8h h1 = *(const v8h*)(hi + base + 16);
            v8h l0 = *(const v8h*)(lo + base);
            v8h l1 = *(const v8h*)(lo + base + 16);
            v16h ah, al;
            #pragma unroll
            for (int t = 0; t < 8; ++t) {
                ah[t] = h0[t]; ah[t + 8] = h1[t];
                al[t] = l0[t]; al[t + 8] = l1[t];
            }
            a_hi[c] = ah; a_lo[c] = al;
        }
    }

    float acc[8];
    #pragma unroll
    for (int r = 0; r < 8; ++r) acc[r] = 0.0f;

    const int ntiles = (jt0 + jchunk <= n_jtiles) ? jchunk : (n_jtiles - jt0);

    // prime the double-buffer pipeline
    if (ntiles > 0) stage_tile(hi, lo, (jt0 + 0) * 16, smem[0], tid);
    if (ntiles > 1) stage_tile(hi, lo, (jt0 + 1) * 16, smem[1], tid);

    for (int t = 0; t < ntiles; ++t) {
        // outstanding async groups: tile t (maybe done) + tile t+1 (if staged).
        // completion is in-order, so <=2 proves tile t has landed.
        if (t + 1 < ntiles) __builtin_amdgcn_s_wait_asynccnt(2);
        else                __builtin_amdgcn_s_wait_asynccnt(0);
        __syncthreads();                       // tile t visible to all waves

        const _Float16* shi = &smem[t & 1][0];
        const _Float16* slo = &smem[t & 1][TILE_H];
        const int boff = half_lane * TROW;

        v8f c_acc = {};
        #pragma unroll
        for (int c = 0; c < 4; ++c) {
            const int bb = boff + c * 32 + hsel * 8;
            v8h bh0 = *(const v8h*)(shi + bb);
            v8h bh1 = *(const v8h*)(shi + bb + 16);
            v8h bl0 = *(const v8h*)(slo + bb);
            v8h bl1 = *(const v8h*)(slo + bb + 16);
            v16h b_hi, b_lo;
            #pragma unroll
            for (int q = 0; q < 8; ++q) {
                b_hi[q] = bh0[q]; b_hi[q + 8] = bh1[q];
                b_lo[q] = bl0[q]; b_lo[q + 8] = bl1[q];
            }
            // hi*hi + hi*lo + lo*hi  (drop lo*lo)
            c_acc = __builtin_amdgcn_wmma_f32_16x16x32_f16(false, a_hi[c], false, b_hi,
                                                           (short)0, c_acc, false, false);
            c_acc = __builtin_amdgcn_wmma_f32_16x16x32_f16(false, a_hi[c], false, b_lo,
                                                           (short)0, c_acc, false, false);
            c_acc = __builtin_amdgcn_wmma_f32_16x16x32_f16(false, a_lo[c], false, b_hi,
                                                           (short)0, c_acc, false, false);
        }

        // c_acc[r] = dot(v_{i0 + r + 8*hsel}, v_{j0 + half_lane})
        #pragma unroll
        for (int r = 0; r < 8; ++r)
            acc[r] += __expf(c_acc[r] * INV_TAU);

        __syncthreads();                       // all waves done reading buf[t&1]
        if (t + 2 < ntiles)                    // overwrite it with tile t+2
            stage_tile(hi, lo, (jt0 + t + 2) * 16, smem[t & 1], tid);
    }

    // reduce over columns: sum across each 16-lane half (xor 1,2,4,8 stays in-half)
    float rsum[8];
    #pragma unroll
    for (int r = 0; r < 8; ++r) {
        float s = acc[r];
        #pragma unroll
        for (int off = 1; off < 16; off <<= 1)
            s += __shfl_xor(s, off, 32);
        rsum[r] = s;
    }

    if (half_lane == 0) {
        const int mbase = i0 + hsel * 8;
        #pragma unroll
        for (int r = 0; r < 8; ++r)
            atomicAdd(&den[mbase + r], rsum[r]);
    }
}

// ---------------------------------------------------------------------------
// 4) finalize: -(1/N) * sum_i ( ||v_i||^2/tau - log(den_i) )
// ---------------------------------------------------------------------------
__global__ __launch_bounds__(256) void id_final_kernel(
    const float* __restrict__ v, const float* __restrict__ den,
    float* __restrict__ out, int n) {

    int i = blockIdx.x * blockDim.x + threadIdx.x;
    float contrib = 0.0f;
    if (i < n) {
        const float4* row = (const float4*)(v + (long)i * KDIM);
        float norm2 = 0.0f;
        #pragma unroll
        for (int t = 0; t < KDIM / 4; ++t) {
            float4 x = row[t];
            norm2 += x.x * x.x + x.y * x.y + x.z * x.z + x.w * x.w;
        }
        float term = norm2 * INV_TAU - __logf(den[i]);
        contrib = -term / (float)n;
    }
    #pragma unroll
    for (int off = 16; off > 0; off >>= 1)
        contrib += __shfl_down(contrib, off, 32);
    if ((threadIdx.x & 31) == 0)
        atomicAdd(out, contrib);
}

// ---------------------------------------------------------------------------
extern "C" void kernel_launch(void* const* d_in, const int* in_sizes, int n_in,
                              void* d_out, int out_size, void* d_ws, size_t ws_size,
                              hipStream_t stream) {
    const float* v = (const float*)d_in[0];
    float* out = (float*)d_out;

    const int total = in_sizes[0];          // N * 128
    const int N = total / KDIM;             // 16384

    // workspace layout: hi[N*K] f16 | lo[N*K] f16 | den[N] f32
    _Float16* hi  = (_Float16*)d_ws;
    _Float16* lo  = hi + (size_t)N * KDIM;
    float*    den = (float*)(lo + (size_t)N * KDIM);

    id_zero_kernel<<<(N + 255) / 256, 256, 0, stream>>>(den, out, N);
    id_split_kernel<<<(total + 255) / 256, 256, 0, stream>>>(v, hi, lo, total);

    const int n_jtiles = N / 16;                     // 1024
    const int n_chunks = 16;                         // parallelism across columns
    const int jchunk   = (n_jtiles + n_chunks - 1) / n_chunks;
    dim3 grid(N / ROWS_PB, n_chunks);                // (128, 16), 256-thread blocks
    id_gram_kernel<<<grid, 256, 0, stream>>>(hi, lo, den, n_jtiles, jchunk);

    id_final_kernel<<<(N + 255) / 256, 256, 0, stream>>>(v, den, out, N);
}
